// MultiModalTransformer_12713103196199
// MI455X (gfx1250) — matrix operand order repaired and checked by reading, not compiled, not run on hardware
//
#include <hip/hip_runtime.h>

// ---------------- constants ----------------
constexpr int Bc = 4, Sc = 512, Dc = 512, Hc = 8, HDc = 64, Lc = 6, QDc = 16;
constexpr int BSc = Bc * Sc;                 // 2048 tokens
constexpr long ATTN1 = (long)Bc * Hc * Sc * Sc;   // 8388608 per layer

typedef __bf16 bf16_t;
typedef __attribute__((ext_vector_type(16))) __bf16 v16bf;
typedef __attribute__((ext_vector_type(8)))  __bf16 v8bf;
typedef __attribute__((ext_vector_type(8)))  float  v8f;

__device__ __forceinline__ bf16_t f2bf(float f) {
  union { float f; unsigned u; } v; v.f = f;
  unsigned r = (v.u + 0x7fffu + ((v.u >> 16) & 1u)) >> 16;
  union { unsigned short s; bf16_t b; } o; o.s = (unsigned short)r;
  return o.b;
}

// ------------- WMMA fragment loaders (ISA 7.12.2 layouts, wave32) -------------
// A 16x32 bf16: lanes 0-15 row M=lane hold K {0..7,16..23}; lanes 16-31 hold K {8..15,24..31}
__device__ __forceinline__ v16bf load_a_bf(const bf16_t* A, int lda, int m0, int k0, int lane) {
  int r = lane & 15, kh = (lane >> 4) * 8;
  const bf16_t* p = A + (long)(m0 + r) * lda + k0 + kh;
  v8bf lo = *(const v8bf*)p;
  v8bf hi = *(const v8bf*)(p + 16);
  v16bf f;
#pragma unroll
  for (int i = 0; i < 8; i++) { f[i] = lo[i]; f[i + 8] = hi[i]; }
  return f;
}
// Same A layout but source is f32 (attention probs), convert on the fly.
__device__ __forceinline__ v16bf load_a_f32(const float* A, int lda, int m0, int k0, int lane) {
  int r = lane & 15, kh = (lane >> 4) * 8;
  const float* p = A + (long)(m0 + r) * lda + k0 + kh;
  v16bf f;
#pragma unroll
  for (int i = 0; i < 8; i++) { f[i] = f2bf(p[i]); f[i + 8] = f2bf(p[16 + i]); }
  return f;
}
// B 32x16 bf16 from N-major (transposed) storage Bt[n*ldb+k]:
// lanes 0-15 col N=lane hold K {0..7,16..23}; lanes 16-31 hold K {8..15,24..31}
__device__ __forceinline__ v16bf load_bt_bf(const bf16_t* Bt, int ldb, int n0, int k0, int lane) {
  int c = lane & 15, kh = (lane >> 4) * 8;
  const bf16_t* p = Bt + (long)(n0 + c) * ldb + k0 + kh;
  v8bf lo = *(const v8bf*)p;
  v8bf hi = *(const v8bf*)(p + 16);
  v16bf f;
#pragma unroll
  for (int i = 0; i < 8; i++) { f[i] = lo[i]; f[i + 8] = hi[i]; }
  return f;
}

__device__ __forceinline__ void store_tile(float* C, long ldc, int m0, int n0, int lane,
                                           const v8f& acc, const float* bias, int act) {
  int cn = n0 + (lane & 15), cm = m0 + (lane >> 4) * 8;
  float bvv = bias ? bias[cn] : 0.0f;
#pragma unroll
  for (int v = 0; v < 8; v++) {
    float r = acc[v] + bvv;
    if (act) r = tanhf(r);
    C[(long)(cm + v) * ldc + cn] = r;
  }
}

#define WMMA_BF16(a, b, c) \
  __builtin_amdgcn_wmma_f32_16x16x32_bf16(false, (a), false, (b), (short)0, (c), false, false)

// ---------------- 2x2 register-tiled bf16 WMMA GEMM ----------------
// C[M,N] = act( A[M,K](bf16) @ Bt[N,K]^T(bf16) + bias[N] ), f32 accumulate
// 256 threads = 8 waves; wave tile 32x32 (4 independent WMMA chains); WG tile 64(M) x 128(N)
__global__ void k_gemm32(const bf16_t* __restrict__ A, const bf16_t* __restrict__ Bt,
                         const float* __restrict__ bias, float* __restrict__ C,
                         int M, int N, int K, int act) {
  int lane = threadIdx.x & 31, w = threadIdx.x >> 5;
  int m0 = blockIdx.y * 64 + (w >> 2) * 32;
  int n0 = blockIdx.x * 128 + (w & 3) * 32;
  if (m0 >= M || n0 >= N) return;       // wave-uniform: EXEC stays all ones
  v8f c00, c01, c10, c11;
  for (int i = 0; i < 8; i++) { c00[i] = 0.f; c01[i] = 0.f; c10[i] = 0.f; c11[i] = 0.f; }
  for (int k0 = 0; k0 < K; k0 += 32) {
    v16bf a0 = load_a_bf(A, K, m0, k0, lane);
    v16bf a1 = load_a_bf(A, K, m0 + 16, k0, lane);
    v16bf b0 = load_bt_bf(Bt, K, n0, k0, lane);
    v16bf b1 = load_bt_bf(Bt, K, n0 + 16, k0, lane);
    // L2-resident stream: hint the next K block (global_prefetch_b8)
    if (k0 + 64 < K) {
      __builtin_prefetch(A + (long)(m0 + (lane & 15)) * K + k0 + 64, 0, 0);
      __builtin_prefetch(Bt + (long)(n0 + (lane & 15)) * K + k0 + 64, 0, 0);
    }
    c00 = WMMA_BF16(a0, b0, c00);
    c01 = WMMA_BF16(a0, b1, c01);
    c10 = WMMA_BF16(a1, b0, c10);
    c11 = WMMA_BF16(a1, b1, c11);
  }
  store_tile(C, N, m0,      n0,      lane, c00, bias, act);
  store_tile(C, N, m0,      n0 + 16, lane, c01, bias, act);
  store_tile(C, N, m0 + 16, n0,      lane, c10, bias, act);
  store_tile(C, N, m0 + 16, n0 + 16, lane, c11, bias, act);
}

// Single-tile GEMM kept for narrow N (the N=16 superposition projections).
__global__ void k_gemm16(const bf16_t* __restrict__ A, const bf16_t* __restrict__ Bt,
                         const float* __restrict__ bias, float* __restrict__ C,
                         int M, int N, int K, int act) {
  int lane = threadIdx.x & 31, w = threadIdx.x >> 5;
  int m0 = blockIdx.y * 32 + (w >> 2) * 16;
  int n0 = blockIdx.x * 64 + (w & 3) * 16;
  if (m0 >= M || n0 >= N) return;
  v8f acc; for (int i = 0; i < 8; i++) acc[i] = 0.0f;
  for (int k0 = 0; k0 < K; k0 += 32) {
    v16bf a  = load_a_bf(A, K, m0, k0, lane);
    v16bf bb = load_bt_bf(Bt, K, n0, k0, lane);
    acc = WMMA_BF16(a, bb, acc);
  }
  store_tile(C, N, m0, n0, lane, acc, bias, act);
}

// scores[bh,i,j] = 0.125 * sum_d Qf[b,i,hd]*Kf[b,j,hd] + 0.1*interf[b,i,j]
// wave tile 32x32, WG 64x128 over the SxS score matrix
__global__ void k_scores(const bf16_t* __restrict__ Qf, const bf16_t* __restrict__ Kf,
                         const float* __restrict__ interf, float* __restrict__ outAttn) {
  int lane = threadIdx.x & 31, w = threadIdx.x >> 5;
  int bh = blockIdx.z, b = bh >> 3, h = bh & 7;
  int m0 = blockIdx.y * 64 + (w >> 2) * 32;
  int n0 = blockIdx.x * 128 + (w & 3) * 32;
  const bf16_t* Abase = Qf + (long)b * Sc * Dc + h * HDc;
  const bf16_t* Bbase = Kf + (long)b * Sc * Dc + h * HDc;
  v8f c00, c01, c10, c11;
  for (int i = 0; i < 8; i++) { c00[i] = 0.f; c01[i] = 0.f; c10[i] = 0.f; c11[i] = 0.f; }
#pragma unroll
  for (int k0 = 0; k0 < HDc; k0 += 32) {
    v16bf a0 = load_a_bf(Abase, Dc, m0, k0, lane);
    v16bf a1 = load_a_bf(Abase, Dc, m0 + 16, k0, lane);
    v16bf b0 = load_bt_bf(Bbase, Dc, n0, k0, lane);
    v16bf b1 = load_bt_bf(Bbase, Dc, n0 + 16, k0, lane);
    c00 = WMMA_BF16(a0, b0, c00);
    c01 = WMMA_BF16(a0, b1, c01);
    c10 = WMMA_BF16(a1, b0, c10);
    c11 = WMMA_BF16(a1, b1, c11);
  }
  float* orow = outAttn + (long)bh * Sc * Sc;
  const float* irow = interf + (long)b * Sc * Sc;
  int cn = n0 + (lane & 15), cm = m0 + (lane >> 4) * 8;
#pragma unroll
  for (int v = 0; v < 8; v++) {
    orow[(long)(cm + v) * Sc + cn]           = c00[v] * 0.125f + 0.1f * irow[(long)(cm + v) * Sc + cn];
    orow[(long)(cm + v) * Sc + cn + 16]      = c01[v] * 0.125f + 0.1f * irow[(long)(cm + v) * Sc + cn + 16];
    orow[(long)(cm + 16 + v) * Sc + cn]      = c10[v] * 0.125f + 0.1f * irow[(long)(cm + 16 + v) * Sc + cn];
    orow[(long)(cm + 16 + v) * Sc + cn + 16] = c11[v] * 0.125f + 0.1f * irow[(long)(cm + 16 + v) * Sc + cn + 16];
  }
}

// ctx[b,i,h*64+d] = sum_j attn[bh,i,j] * Vt[b, h*64+d, j]
// wave tile 32x32, 8 waves arranged 4(M) x 2(N) -> WG 128x64
__global__ void k_ctx(const float* __restrict__ attn, const bf16_t* __restrict__ Vt,
                      float* __restrict__ ctx) {
  int lane = threadIdx.x & 31, w = threadIdx.x >> 5;
  int bh = blockIdx.z, b = bh >> 3, h = bh & 7;
  int m0 = blockIdx.y * 128 + (w >> 1) * 32;
  int n0 = (w & 1) * 32;    // N = 64
  const float*  Abase = attn + (long)bh * Sc * Sc;
  const bf16_t* Bbase = Vt + (long)b * Dc * Sc + (long)(h * HDc) * Sc;
  v8f c00, c01, c10, c11;
  for (int i = 0; i < 8; i++) { c00[i] = 0.f; c01[i] = 0.f; c10[i] = 0.f; c11[i] = 0.f; }
  for (int k0 = 0; k0 < Sc; k0 += 32) {
    v16bf a0 = load_a_f32(Abase, Sc, m0, k0, lane);
    v16bf a1 = load_a_f32(Abase, Sc, m0 + 16, k0, lane);
    v16bf b0 = load_bt_bf(Bbase, Sc, n0, k0, lane);
    v16bf b1 = load_bt_bf(Bbase, Sc, n0 + 16, k0, lane);
    c00 = WMMA_BF16(a0, b0, c00);
    c01 = WMMA_BF16(a0, b1, c01);
    c10 = WMMA_BF16(a1, b0, c10);
    c11 = WMMA_BF16(a1, b1, c11);
  }
  int cn = n0 + (lane & 15), cm = m0 + (lane >> 4) * 8;
  float* obase = ctx + (long)b * Sc * Dc + h * HDc;
#pragma unroll
  for (int v = 0; v < 8; v++) {
    obase[(long)(cm + v) * Dc + cn]           = c00[v];
    obase[(long)(cm + v) * Dc + cn + 16]      = c01[v];
    obase[(long)(cm + 16 + v) * Dc + cn]      = c10[v];
    obase[(long)(cm + 16 + v) * Dc + cn + 16] = c11[v];
  }
}

// ---------------- elementwise / small kernels ----------------
__global__ void k_transpose_bf(const float* __restrict__ in, bf16_t* __restrict__ out,
                               int rows, int cols) {
  int t = blockIdx.x * blockDim.x + threadIdx.x;
  if (t >= rows * cols) return;
  int r = t / cols, c = t % cols;
  out[(long)c * rows + r] = f2bf(in[t]);
}

__global__ void k_f2bf(const float* __restrict__ in, bf16_t* __restrict__ out, int n) {
  int t = blockIdx.x * blockDim.x + threadIdx.x;
  if (t < n) out[t] = f2bf(in[t]);
}

__global__ void k_copy(const float* __restrict__ in, float* __restrict__ out, int n) {
  int t = blockIdx.x * blockDim.x + threadIdx.x;
  if (t < n) out[t] = in[t];
}

__global__ void k_modal(const float* __restrict__ visual, const float* __restrict__ Wvis,
                        const float* __restrict__ bvis, const float* __restrict__ temporal,
                        const float* __restrict__ Wtemp, const float* __restrict__ btemp,
                        float* __restrict__ visproj, float* __restrict__ tempproj) {
  int i = blockIdx.x * blockDim.x + threadIdx.x;
  if (i >= Bc * Dc) return;
  int b = i / Dc, d = i % Dc;
  float a = bvis[d];
  for (int k = 0; k < 2048; k++) a += visual[b * 2048 + k] * Wvis[(long)k * Dc + d];
  visproj[i] = a;
  float c = btemp[d];
  for (int k = 0; k < 64; k++) c += temporal[b * 64 + k] * Wtemp[k * Dc + d];
  tempproj[i] = c;
}

__global__ void k_buildx(const int* __restrict__ text, const float* __restrict__ embed,
                         const float* __restrict__ visproj, const float* __restrict__ tempproj,
                         float* __restrict__ x) {
  int t = blockIdx.x;                 // token 0..BS-1
  int b = t / Sc, s = t % Sc;
  int id = text[t];
  for (int d = threadIdx.x; d < Dc; d += blockDim.x) {
    int i = d & ~1;
    float div = __expf(-9.210340371976184f * (float)i * (1.0f / 512.0f));
    float ang = (float)s * div;
    float pe = (d & 1) ? __cosf(ang) : __sinf(ang);
    x[(long)t * Dc + d] = embed[(long)id * Dc + d] + visproj[b * Dc + d] + tempproj[b * Dc + d] + pe;
  }
}

// Quantum tail: ev = qs@qwr/qwi ; en = ev@em ; probs = |en|^2 ; qout = probs@mb
// fw = sigmoid(mean(probs)) ; fused = 0.7*Lin + 0.3*(fw*qout + (1-fw)*Classical) -> bf16 frag
__global__ void k_qtail(const float* __restrict__ qs, const float* __restrict__ qwr,
                        const float* __restrict__ qwi, const float* __restrict__ em,
                        const float* __restrict__ mb, const float* __restrict__ Cl,
                        const float* __restrict__ Lin, float* __restrict__ probs_out,
                        bf16_t* __restrict__ frag_out) {
  __shared__ float sqs[QDc], sevr[QDc], sevi[QDc], spr[QDc], sfw;
  int t = blockIdx.x, tid = threadIdx.x;
  if (tid < QDc) sqs[tid] = qs[t * QDc + tid];
  __syncthreads();
  if (tid < QDc) {
    float er = 0.0f, ei = 0.0f;
    for (int p = 0; p < QDc; p++) { float q = sqs[p]; er += q * qwr[p * QDc + tid]; ei += q * qwi[p * QDc + tid]; }
    sevr[tid] = er; sevi[tid] = ei;
  }
  __syncthreads();
  if (tid < QDc) {
    float nr = 0.0f, ni = 0.0f;
    for (int p = 0; p < QDc; p++) { float e = em[p * QDc + tid]; nr += sevr[p] * e; ni += sevi[p] * e; }
    float pr = nr * nr + ni * ni;
    spr[tid] = pr;
    probs_out[t * QDc + tid] = pr;
  }
  __syncthreads();
  if (tid == 0) {
    float s = 0.0f;
    for (int q = 0; q < QDc; q++) s += spr[q];
    s *= (1.0f / QDc);
    sfw = 1.0f / (1.0f + __expf(-s));
  }
  __syncthreads();
  float fw = sfw;
  for (int d = tid; d < Dc; d += blockDim.x) {
    float qo = 0.0f;
    for (int q = 0; q < QDc; q++) qo += spr[q] * mb[q * Dc + d];
    float qv = fw * qo + (1.0f - fw) * Cl[(long)t * Dc + d];
    frag_out[(long)t * Dc + d] = f2bf(0.7f * Lin[(long)t * Dc + d] + 0.3f * qv);
  }
}

__global__ void k_interf(const float* __restrict__ qp, const float* __restrict__ kp,
                         float* __restrict__ interf) {
  int j = blockIdx.x * blockDim.x + threadIdx.x;
  int i = blockIdx.y, b = blockIdx.z;
  const float* qr = qp + (long)(b * Sc + i) * QDc;
  const float* kr = kp + (long)(b * Sc + j) * QDc;
  float acc = 0.0f;
#pragma unroll
  for (int q = 0; q < QDc; q++) acc += qr[q] * kr[q];
  interf[((long)b * Sc + i) * Sc + j] = acc;
}

__global__ void k_softmax(float* __restrict__ base) {
  __shared__ float sd[256];
  float* p = base + (long)blockIdx.x * Sc;
  int t = threadIdx.x;
  float a = p[t], b = p[t + 256];
  sd[t] = fmaxf(a, b);
  __syncthreads();
  for (int s = 128; s > 0; s >>= 1) { if (t < s) sd[t] = fmaxf(sd[t], sd[t + s]); __syncthreads(); }
  float mx = sd[0];
  __syncthreads();
  float ea = __expf(a - mx), eb = __expf(b - mx);
  sd[t] = ea + eb;
  __syncthreads();
  for (int s = 128; s > 0; s >>= 1) { if (t < s) sd[t] += sd[t + s]; __syncthreads(); }
  float inv = 1.0f / sd[0];
  p[t] = ea * inv; p[t + 256] = eb * inv;
}

__global__ void k_ln(const float* __restrict__ yproj, float* __restrict__ x,
                     const float* __restrict__ g, const float* __restrict__ bt) {
  __shared__ float sd[256];
  int tid = threadIdx.x;
  const float* yp = yproj + (long)blockIdx.x * Dc;
  float* xp = x + (long)blockIdx.x * Dc;
  float y0 = yp[tid] + xp[tid];
  float y1 = yp[tid + 256] + xp[tid + 256];
  sd[tid] = y0 + y1;
  __syncthreads();
  for (int s = 128; s > 0; s >>= 1) { if (tid < s) sd[tid] += sd[tid + s]; __syncthreads(); }
  float mu = sd[0] * (1.0f / Dc);
  __syncthreads();
  float d0 = y0 - mu, d1 = y1 - mu;
  sd[tid] = d0 * d0 + d1 * d1;
  __syncthreads();
  for (int s = 128; s > 0; s >>= 1) { if (tid < s) sd[tid] += sd[tid + s]; __syncthreads(); }
  float rs = rsqrtf(sd[0] * (1.0f / Dc) + 1e-5f);
  __syncthreads();
  xp[tid]       = d0 * rs * g[tid] + bt[tid];
  xp[tid + 256] = d1 * rs * g[tid + 256] + bt[tid + 256];
}

__global__ void k_pool(const float* __restrict__ x, float* __restrict__ pooled) {
  int i = blockIdx.x * blockDim.x + threadIdx.x;
  if (i >= Bc * Dc) return;
  int b = i / Dc, d = i % Dc;
  float acc = 0.0f;
  for (int s = 0; s < Sc; s++) acc += x[((long)b * Sc + s) * Dc + d];
  pooled[i] = acc * (1.0f / Sc);
}

__global__ void k_heads(const float* __restrict__ pooled, const float* __restrict__ Ws,
                        const float* __restrict__ bs_, const float* __restrict__ Wd,
                        const float* __restrict__ bd, const float* __restrict__ We,
                        const float* __restrict__ be, float* __restrict__ out) {
  __shared__ float slog[5];
  int b = blockIdx.x, t = threadIdx.x;
  const float* p = pooled + b * Dc;
  if (t < 64) {
    float acc = bs_[t];
    for (int d = 0; d < Dc; d++) acc += p[d] * Ws[d * 64 + t];
    out[b * 64 + t] = 1.0f / (1.0f + __expf(-acc));
  } else if (t < 69) {
    int j = t - 64;
    float acc = bd[j];
    for (int d = 0; d < Dc; d++) acc += p[d] * Wd[d * 5 + j];
    slog[j] = acc;
  } else if (t == 69) {
    float acc = be[0];
    for (int d = 0; d < Dc; d++) acc += p[d] * We[d];
    out[276 + b] = 1.0f / (1.0f + __expf(-acc));
  }
  __syncthreads();
  if (t == 0) {
    float mx = slog[0];
    for (int j = 1; j < 5; j++) mx = fmaxf(mx, slog[j]);
    float s = 0.0f, e[5];
    for (int j = 0; j < 5; j++) { e[j] = __expf(slog[j] - mx); s += e[j]; }
    for (int j = 0; j < 5; j++) out[256 + b * 5 + j] = e[j] / s;
  }
}

// ---------------- host orchestration ----------------
extern "C" void kernel_launch(void* const* d_in, const int* in_sizes, int n_in,
                              void* d_out, int out_size, void* d_ws, size_t ws_size,
                              hipStream_t stream) {
  (void)in_sizes; (void)n_in; (void)out_size; (void)ws_size;
  const int*   text     = (const int*)  d_in[0];
  const float* visual   = (const float*)d_in[1];
  const float* temporal = (const float*)d_in[2];
  const float* embed    = (const float*)d_in[3];
  const float* Wvis     = (const float*)d_in[4];
  const float* bvis     = (const float*)d_in[5];
  const float* Wtemp    = (const float*)d_in[6];
  const float* btemp    = (const float*)d_in[7];
  const float* Wq = (const float*)d_in[8];   const float* bq = (const float*)d_in[9];
  const float* Wk = (const float*)d_in[10];  const float* bk = (const float*)d_in[11];
  const float* Wv = (const float*)d_in[12];  const float* bv = (const float*)d_in[13];
  const float* Wo = (const float*)d_in[14];  const float* bo = (const float*)d_in[15];
  const float* qW = (const float*)d_in[16];  const float* qb = (const float*)d_in[17];
  const float* qwr = (const float*)d_in[18]; const float* qwi = (const float*)d_in[19];
  const float* qsg = (const float*)d_in[20]; const float* qem = (const float*)d_in[21];
  const float* qmb = (const float*)d_in[22];
  const float* kW = (const float*)d_in[23];  const float* kb = (const float*)d_in[24];
  const float* kwr = (const float*)d_in[25]; const float* kwi = (const float*)d_in[26];
  const float* ksg = (const float*)d_in[27]; const float* kem = (const float*)d_in[28];
  const float* kmb = (const float*)d_in[29];
  const float* lng = (const float*)d_in[30]; const float* lnb = (const float*)d_in[31];
  const float* Ws  = (const float*)d_in[32]; const float* bs_ = (const float*)d_in[33];
  const float* Wd  = (const float*)d_in[34]; const float* bd  = (const float*)d_in[35];
  const float* We  = (const float*)d_in[36]; const float* be  = (const float*)d_in[37];
  float* out = (float*)d_out;

  // ---- workspace carve ----
  char* wsp = (char*)d_ws; size_t off = 0;
  auto carve = [&](size_t bytes) -> void* {
    void* p = wsp + off; off += (bytes + 255) & ~(size_t)255; return p;
  };
  const size_t DDsz = (size_t)Dc * Dc;
  bf16_t* wtq  = (bf16_t*)carve((size_t)Lc * DDsz * 2);
  bf16_t* wtk  = (bf16_t*)carve((size_t)Lc * DDsz * 2);
  bf16_t* wtv  = (bf16_t*)carve((size_t)Lc * DDsz * 2);
  bf16_t* wtqW = (bf16_t*)carve((size_t)Lc * DDsz * 2);
  bf16_t* wtkW = (bf16_t*)carve((size_t)Lc * DDsz * 2);
  bf16_t* wto  = (bf16_t*)carve((size_t)Lc * DDsz * 2);
  bf16_t* sgqt = (bf16_t*)carve((size_t)Lc * QDc * Dc * 2);
  bf16_t* sgkt = (bf16_t*)carve((size_t)Lc * QDc * Dc * 2);
  bf16_t* xb   = (bf16_t*)carve((size_t)BSc * Dc * 2);
  bf16_t* Qfb  = (bf16_t*)carve((size_t)BSc * Dc * 2);
  bf16_t* Kfb  = (bf16_t*)carve((size_t)BSc * Dc * 2);
  bf16_t* Vtb  = (bf16_t*)carve((size_t)Bc * Dc * Sc * 2);
  bf16_t* ctxb = (bf16_t*)carve((size_t)BSc * Dc * 2);
  float* Qb   = (float*)carve((size_t)BSc * Dc * 4);
  float* Kb   = (float*)carve((size_t)BSc * Dc * 4);
  float* Vb   = (float*)carve((size_t)BSc * Dc * 4);
  float* Cq   = (float*)carve((size_t)BSc * Dc * 4);
  float* Ck   = (float*)carve((size_t)BSc * Dc * 4);
  float* qsq  = (float*)carve((size_t)BSc * QDc * 4);
  float* qsk  = (float*)carve((size_t)BSc * QDc * 4);
  float* qpb  = (float*)carve((size_t)BSc * QDc * 4);
  float* kpb  = (float*)carve((size_t)BSc * QDc * 4);
  float* interf = (float*)carve((size_t)Bc * Sc * Sc * 4);
  float* ctx  = (float*)carve((size_t)BSc * Dc * 4);
  float* yproj = (float*)carve((size_t)BSc * Dc * 4);
  float* xbuf  = (float*)carve((size_t)BSc * Dc * 4);
  float* visproj  = (float*)carve((size_t)Bc * Dc * 4);
  float* tempproj = (float*)carve((size_t)Bc * Dc * 4);
  float* pooled   = (float*)carve((size_t)Bc * Dc * 4);

  const int nDD = Dc * Dc, gDD = (nDD + 255) / 256;
  const int nSG = Dc * QDc, gSG = (nSG + 255) / 256;
  const int nX = BSc * Dc, gX = (nX + 255) / 256;

  // ---- weight prep: transpose + bf16 convert (weights stay L2-resident) ----
  for (int l = 0; l < Lc; l++) {
    size_t o = (size_t)l * DDsz;
    k_transpose_bf<<<gDD, 256, 0, stream>>>(Wq + o, wtq + o, Dc, Dc);
    k_transpose_bf<<<gDD, 256, 0, stream>>>(Wk + o, wtk + o, Dc, Dc);
    k_transpose_bf<<<gDD, 256, 0, stream>>>(Wv + o, wtv + o, Dc, Dc);
    k_transpose_bf<<<gDD, 256, 0, stream>>>(qW + o, wtqW + o, Dc, Dc);
    k_transpose_bf<<<gDD, 256, 0, stream>>>(kW + o, wtkW + o, Dc, Dc);
    k_transpose_bf<<<gDD, 256, 0, stream>>>(Wo + o, wto + o, Dc, Dc);
    k_transpose_bf<<<gSG, 256, 0, stream>>>(qsg + (size_t)l * Dc * QDc, sgqt + (size_t)l * QDc * Dc, Dc, QDc);
    k_transpose_bf<<<gSG, 256, 0, stream>>>(ksg + (size_t)l * Dc * QDc, sgkt + (size_t)l * QDc * Dc, Dc, QDc);
  }

  // ---- embeddings + modal projections + positional encodings ----
  k_modal<<<(Bc * Dc + 255) / 256, 256, 0, stream>>>(visual, Wvis, bvis, temporal, Wtemp, btemp,
                                                     visproj, tempproj);
  k_buildx<<<BSc, 256, 0, stream>>>(text, embed, visproj, tempproj, xbuf);

  float* attnsBase = out + 280 + (size_t)BSc * Dc;

  for (int l = 0; l < Lc; l++) {
    size_t lDD = (size_t)l * DDsz;
    size_t lSG = (size_t)l * QDc * Dc;
    size_t lQQ = (size_t)l * QDc * QDc;
    float* attnL = attnsBase + (size_t)l * ATTN1;

    k_f2bf<<<gX, 256, 0, stream>>>(xbuf, xb, nX);

    dim3 gg((Dc + 127) / 128, (BSc + 63) / 64);   // 2x2 register-tiled GEMM grid
    k_gemm32<<<gg, 256, 0, stream>>>(xb, wtq + lDD, bq + l * Dc, Qb, BSc, Dc, Dc, 0);
    k_gemm32<<<gg, 256, 0, stream>>>(xb, wtk + lDD, bk + l * Dc, Kb, BSc, Dc, Dc, 0);
    k_gemm32<<<gg, 256, 0, stream>>>(xb, wtv + lDD, bv + l * Dc, Vb, BSc, Dc, Dc, 0);
    k_gemm32<<<gg, 256, 0, stream>>>(xb, wtqW + lDD, qb + l * Dc, Cq, BSc, Dc, Dc, 1);  // tanh
    k_gemm32<<<gg, 256, 0, stream>>>(xb, wtkW + lDD, kb + l * Dc, Ck, BSc, Dc, Dc, 1);  // tanh

    dim3 gq(1, (BSc + 31) / 32);
    k_gemm16<<<gq, 256, 0, stream>>>(xb, sgqt + lSG, nullptr, qsq, BSc, QDc, Dc, 0);
    k_gemm16<<<gq, 256, 0, stream>>>(xb, sgkt + lSG, nullptr, qsk, BSc, QDc, Dc, 0);

    k_qtail<<<BSc, 128, 0, stream>>>(qsq, qwr + lQQ, qwi + lQQ, qem + lQQ, qmb + lSG,
                                     Cq, Qb, qpb, Qfb);
    k_qtail<<<BSc, 128, 0, stream>>>(qsk, kwr + lQQ, kwi + lQQ, kem + lQQ, kmb + lSG,
                                     Ck, Kb, kpb, Kfb);

    k_interf<<<dim3(Sc / 256, Sc, Bc), 256, 0, stream>>>(qpb, kpb, interf);

    k_scores<<<dim3(Sc / 128, Sc / 64, Bc * Hc), 256, 0, stream>>>(Qfb, Kfb, interf, attnL);
    k_softmax<<<Bc * Hc * Sc, 256, 0, stream>>>(attnL);

    for (int b2 = 0; b2 < Bc; b2++)
      k_transpose_bf<<<(Sc * Dc + 255) / 256, 256, 0, stream>>>(
          Vb + (size_t)b2 * Sc * Dc, Vtb + (size_t)b2 * Dc * Sc, Sc, Dc);

    k_ctx<<<dim3(1, Sc / 128, Bc * Hc), 256, 0, stream>>>(attnL, Vtb, ctx);

    k_f2bf<<<gX, 256, 0, stream>>>(ctx, ctxb, nX);
    k_gemm32<<<gg, 256, 0, stream>>>(ctxb, wto + lDD, bo + l * Dc, yproj, BSc, Dc, Dc, 0);
    k_ln<<<BSc, 256, 0, stream>>>(yproj, xbuf, lng + l * Dc, lnb + l * Dc);
  }

  // ---- outputs ----
  k_copy<<<gX, 256, 0, stream>>>(xbuf, out + 280, nX);
  k_pool<<<(Bc * Dc + 255) / 256, 256, 0, stream>>>(xbuf, pooled);
  k_heads<<<Bc, 128, 0, stream>>>(pooled, Ws, bs_, Wd, bd, We, be, out);
}
